// Attention_89472758710727
// MI455X (gfx1250) — compile-verified, matrix operand order checked by reading, not run on hardware
//
#include <hip/hip_runtime.h>

typedef __attribute__((ext_vector_type(16))) _Float16 v16h;
typedef __attribute__((ext_vector_type(8)))  float    v8f;
typedef __attribute__((ext_vector_type(8)))  _Float16 h8;
typedef __attribute__((ext_vector_type(4)))  _Float16 h4;

static constexpr int HEADS = 8;
static constexpr int DHD   = 64;
static constexpr int QBN   = 16;
static constexpr int NSQ   = 512;
static constexpr int DIMX  = 512;
static constexpr size_t FSZ = (size_t)HEADS * QBN * NSQ * DHD;   // 4,194,304 elems per f tensor
static constexpr int MROWS = QBN * NSQ;                          // 8192
static constexpr int TROWS = 3 * MROWS;                          // 24576

__device__ __forceinline__ float clampf(float x, float lo, float hi) {
  return fminf(fmaxf(x, lo), hi);
}

__device__ __forceinline__ v8f wmma_f16(v16h a, v16h b, v8f c) {
  return __builtin_amdgcn_wmma_f32_16x16x32_f16(false, a, false, b, (short)0, c, false, false);
}

// A-fragment (16x32 f16) from row-major [row][k] LDS (stride multiple of 8 halves).
// CDNA5 layout: lanes 0-15 hold row (lane&15), K = {kb..kb+7, kb+16..kb+23}, kb = 0 (lanes<16) / 8.
// Both 8-half runs are contiguous & 16B aligned -> two ds_load_b128 per fragment.
// Also used for B when tile is stored transposed ([N][K]) since that mirrors the A layout.
__device__ __forceinline__ v16h frag_rowmajor(const _Float16* s, int stride, int row0, int k0, int lane) {
  int r  = row0 + (lane & 15);
  int kb = (lane < 16) ? 0 : 8;
  const _Float16* p = s + r * stride + k0 + kb;
  h8 lo = *(const h8*)p;          // K = kb .. kb+7
  h8 hi = *(const h8*)(p + 16);   // K = kb+16 .. kb+23
  return __builtin_shufflevector(lo, hi, 0, 1, 2, 3, 4, 5, 6, 7, 8, 9, 10, 11, 12, 13, 14, 15);
}

// ---------------------------------------------------------------- init / convert
__global__ void k_init(float* feat, float* hsum, float* coef, float* wts) {
  int t = threadIdx.x;
  for (int i = t; i < 1024; i += 256) feat[i] = 0.f;
  if (t < 128) hsum[t] = 0.f;
  if (t < 64)  coef[t] = 0.f;
  if (t < 32)  wts[t]  = 0.f;
}

__global__ void k_cvt(const float* __restrict__ win, const float* __restrict__ wout,
                      _Float16* __restrict__ win16, _Float16* __restrict__ wout16) {
  int i = blockIdx.x * 256 + threadIdx.x;
  if (i < 512 * 512) {
    win16[i]  = (_Float16)win[i];
    wout16[i] = (_Float16)wout[i];
  }
}

// ---------------------------------------------------------------- LayerNorm -> f16
__global__ void k_ln(const float* __restrict__ q, const float* __restrict__ k,
                     const float* __restrict__ v, const float* __restrict__ g,
                     const float* __restrict__ bta, _Float16* __restrict__ xln) {
  int wave = threadIdx.x >> 5, lane = threadIdx.x & 31;
  int row = blockIdx.x * 8 + wave;                     // < 24576
  const float* src = (row < MROWS) ? q : ((row < 2 * MROWS) ? k : v);
  int r = row & (MROWS - 1);
  const float4* p = (const float4*)(src + (size_t)r * DIMX);
  float4 x[4];
  float s = 0.f, ss = 0.f;
#pragma unroll
  for (int j = 0; j < 4; ++j) {
    x[j] = p[lane + 32 * j];
    s  += x[j].x + x[j].y + x[j].z + x[j].w;
    ss += x[j].x * x[j].x + x[j].y * x[j].y + x[j].z * x[j].z + x[j].w * x[j].w;
  }
#pragma unroll
  for (int m = 16; m >= 1; m >>= 1) { s += __shfl_xor(s, m); ss += __shfl_xor(ss, m); }
  float mu   = s * (1.f / 512.f);
  float var  = ss * (1.f / 512.f) - mu * mu;           // biased, matches LayerNorm
  float rstd = rsqrtf(var + 1e-5f);
  _Float16* dst = xln + (size_t)row * DIMX;
#pragma unroll
  for (int j = 0; j < 4; ++j) {
    float4 gg = ((const float4*)g)[lane + 32 * j];
    float4 bb = ((const float4*)bta)[lane + 32 * j];
    h4 o;
    o[0] = (_Float16)((x[j].x - mu) * rstd * gg.x + bb.x);
    o[1] = (_Float16)((x[j].y - mu) * rstd * gg.y + bb.y);
    o[2] = (_Float16)((x[j].z - mu) * rstd * gg.z + bb.z);
    o[3] = (_Float16)((x[j].w - mu) * rstd * gg.w + bb.w);
    ((h4*)dst)[lane + 32 * j] = o;
  }
}

// ---------------------------------------------------------------- proj GEMM: Xln @ W_in -> f[h,qb,n,d] f16
__global__ void k_gemm_in(const _Float16* __restrict__ xln, const _Float16* __restrict__ w16,
                          _Float16* __restrict__ f16buf) {
  __shared__ _Float16 As[128][72];
  __shared__ _Float16 Bst[64][72];    // transposed weight tile: [n][k]
  int tid = threadIdx.x, wave = tid >> 5, lane = tid & 31;
  int wr = wave & 3, wc = wave >> 2;
  int row0 = blockIdx.x * 128, col0 = blockIdx.y * 64;
  v8f acc[2][2];
#pragma unroll
  for (int a = 0; a < 2; ++a)
#pragma unroll
    for (int b = 0; b < 2; ++b)
#pragma unroll
      for (int i = 0; i < 8; ++i) acc[a][b][i] = 0.f;
  for (int kb = 0; kb < 512; kb += 64) {
#pragma unroll
    for (int it = 0; it < 4; ++it) {
      int pos = (tid + it * 256) * 8; int r = pos >> 6, c = pos & 63;
      *(h8*)&As[r][c] = *(const h8*)&xln[(size_t)(row0 + r) * 512 + kb + c];
    }
#pragma unroll
    for (int it = 0; it < 2; ++it) {
      int pos = (tid + it * 256) * 8; int r = pos >> 6, c = pos & 63;   // r = k, c = n-base
      h8 bv = *(const h8*)&w16[(size_t)(kb + r) * 512 + col0 + c];
#pragma unroll
      for (int e = 0; e < 8; ++e) Bst[c + e][r] = bv[e];
    }
    if (kb + 64 < 512)
      __builtin_prefetch(&xln[(size_t)(row0 + (tid >> 1)) * 512 + kb + 64], 0, 1);
    __syncthreads();
#pragma unroll
    for (int kk = 0; kk < 2; ++kk) {
      v16h af[2], bf[2];
#pragma unroll
      for (int rt = 0; rt < 2; ++rt) af[rt] = frag_rowmajor(&As[0][0], 72, wr * 32 + rt * 16, kk * 32, lane);
#pragma unroll
      for (int ct = 0; ct < 2; ++ct) bf[ct] = frag_rowmajor(&Bst[0][0], 72, wc * 32 + ct * 16, kk * 32, lane);
#pragma unroll
      for (int rt = 0; rt < 2; ++rt)
#pragma unroll
        for (int ct = 0; ct < 2; ++ct) acc[rt][ct] = wmma_f16(af[rt], bf[ct], acc[rt][ct]);
    }
    __syncthreads();
  }
  int half = lane >> 4, cl = lane & 15;
#pragma unroll
  for (int rt = 0; rt < 2; ++rt)
#pragma unroll
    for (int ct = 0; ct < 2; ++ct)
#pragma unroll
      for (int v = 0; v < 8; ++v) {
        int rg = row0 + wr * 32 + rt * 16 + half * 8 + v;
        int cg = col0 + wc * 32 + ct * 16 + cl;
        int t = rg >> 13;               // tensor (q/k/v)
        int r = rg & 8191;
        int qb = r >> 9, n = r & 511;
        int hh = cg >> 6, d = cg & 63;
        f16buf[(size_t)t * FSZ + (((size_t)((hh * 16 + qb) * 512 + n)) << 6) + d] = (_Float16)acc[rt][ct][v];
      }
}

// ---------------------------------------------------------------- per-(h,qb) row/col stats + head feats
__global__ void k_stats2(const _Float16* __restrict__ f16buf,
                         float* __restrict__ ksum, float* __restrict__ ik1, float* __restrict__ ik2,
                         float* __restrict__ qm, float* __restrict__ qkm,
                         float* __restrict__ iq1, float* __restrict__ iq2,
                         float* __restrict__ kmsum, float* __restrict__ feat) {
  int hq = blockIdx.x, h = hq >> 4;
  const _Float16* fq = f16buf + (size_t)hq * NSQ * DHD;
  const _Float16* fk = f16buf + FSZ + (size_t)hq * NSQ * DHD;
  __shared__ float km[64];
  __shared__ float colred[4][64];
  int tid = threadIdx.x;
  int d = tid & 63, rb = tid >> 6;
  // fk column sums -> km, featk
  float s = 0.f;
  for (int n = rb; n < 512; n += 4) s += (float)fk[n * 64 + d];
  colred[rb][d] = s; __syncthreads();
  if (tid < 64) {
    float t = colred[0][tid] + colred[1][tid] + colred[2][tid] + colred[3][tid];
    km[tid] = t * (1.f / 512.f);
    atomicAdd(&feat[h * 128 + 64 + tid], t);
  }
  __syncthreads();
  // fq column sums -> featq
  s = 0.f;
  for (int n = rb; n < 512; n += 4) s += (float)fq[n * 64 + d];
  colred[rb][d] = s; __syncthreads();
  if (tid < 64)
    atomicAdd(&feat[h * 128 + tid], colred[0][tid] + colred[1][tid] + colred[2][tid] + colred[3][tid]);
  if (tid == 0) { float t = 0.f; for (int i = 0; i < 64; ++i) t += km[i]; kmsum[hq] = t; }
  __syncthreads();
  // per-row key stats
  for (int r = tid; r < 512; r += 256) {
    float sv = 0.f, sq = 0.f;
    for (int dd = 0; dd < 64; ++dd) { float x = (float)fk[r * 64 + dd]; sv += x; sq += x * x; }
    float nrm = sqrtf(sq);
    ksum[hq * 512 + r] = sv;
    ik1[hq * 512 + r] = 1.f / (nrm + 1e-6f);
    ik2[hq * 512 + r] = 1.f / fmaxf(nrm, 1e-6f);
  }
  // per-row query stats
  for (int r = tid; r < 512; r += 256) {
    float sv = 0.f, sq = 0.f, dk = 0.f;
    for (int dd = 0; dd < 64; ++dd) { float x = (float)fq[r * 64 + dd]; sv += x; sq += x * x; dk += x * km[dd]; }
    float nrm = sqrtf(sq);
    qm[hq * 512 + r]  = sv * (1.f / 64.f);
    qkm[hq * 512 + r] = dk;
    iq1[hq * 512 + r] = 1.f / (nrm + 1e-6f);
    iq2[hq * 512 + r] = 1.f / fmaxf(nrm, 1e-6f);
  }
}

// ---------------------------------------------------------------- score-stat pass (9 per-head moments + varrow)
__global__ void k_scores(const _Float16* __restrict__ f16buf,
                         const float* __restrict__ ksum, const float* __restrict__ ik1, const float* __restrict__ ik2,
                         const float* __restrict__ qm, const float* __restrict__ qkm,
                         const float* __restrict__ iq1, const float* __restrict__ iq2,
                         const float* __restrict__ kmsum, float* __restrict__ varrow,
                         float* __restrict__ hsum) {
  __shared__ _Float16 Qs[128][72];
  __shared__ _Float16 Ks[64][72];      // [m][d] row-major = transposed-B layout for QK^T
  __shared__ float rs_c1[128], rs_cov[128], rs_mar[128];
  __shared__ float red[8][6];
  __shared__ float acc4[4];
  int blk = blockIdx.x, hq = blk >> 2, rb = blk & 3, h = hq >> 4, n0 = rb * 128;
  int tid = threadIdx.x, wave = tid >> 5, lane = tid & 31, half = lane >> 4, cl = lane & 15;
  const _Float16* fq = f16buf + (size_t)hq * NSQ * DHD;
  const _Float16* fk = f16buf + FSZ + (size_t)hq * NSQ * DHD;
#pragma unroll
  for (int it = 0; it < 4; ++it) {
    int pos = (tid + it * 256) * 8; int r = pos >> 6, c = pos & 63;
    *(h8*)&Qs[r][c] = *(const h8*)&fq[(size_t)(n0 + r) * 64 + c];
  }
  if (tid < 128) { rs_c1[tid] = 0.f; rs_cov[tid] = 0.f; rs_mar[tid] = 0.f; }
  if (tid < 4) acc4[tid] = 0.f;
  int rowbase = hq * 512 + n0 + wave * 16 + half * 8;
  float qm_v[8], qkm_v[8], iq1_v[8], iq2_v[8];
#pragma unroll
  for (int v = 0; v < 8; ++v) {
    qm_v[v] = qm[rowbase + v]; qkm_v[v] = qkm[rowbase + v];
    iq1_v[v] = iq1[rowbase + v]; iq2_v[v] = iq2[rowbase + v];
  }
  float kms = kmsum[hq];
  const float covscale = (1e-3f / 512.f) / (8.f + 1e-6f);
  float p0 = 0.f, p1 = 0.f, p2 = 0.f, p3 = 0.f, p4 = 0.f;
  __syncthreads();
  for (int mc = 0; mc < 8; ++mc) {
    int m0 = mc * 64;
#pragma unroll
    for (int it = 0; it < 2; ++it) {
      int pos = (tid + it * 256) * 8; int r = pos >> 6, c = pos & 63;
      *(h8*)&Ks[r][c] = *(const h8*)&fk[(size_t)(m0 + r) * 64 + c];
    }
    __syncthreads();
    float ks_c[4], ik1_c[4], ik2_c[4];
#pragma unroll
    for (int ct = 0; ct < 4; ++ct) {
      int mi = hq * 512 + m0 + ct * 16 + cl;
      ks_c[ct] = ksum[mi]; ik1_c[ct] = ik1[mi]; ik2_c[ct] = ik2[mi];
    }
    v8f acc[4];
#pragma unroll
    for (int ct = 0; ct < 4; ++ct)
#pragma unroll
      for (int i = 0; i < 8; ++i) acc[ct][i] = 0.f;
#pragma unroll
    for (int kk = 0; kk < 2; ++kk) {
      v16h a = frag_rowmajor(&Qs[0][0], 72, wave * 16, kk * 32, lane);
#pragma unroll
      for (int ct = 0; ct < 4; ++ct) {
        v16h b = frag_rowmajor(&Ks[0][0], 72, ct * 16, kk * 32, lane);  // B = fk^T
        acc[ct] = wmma_f16(a, b, acc[ct]);
      }
    }
    float rp1[8], rp2[8], rp3[8];
#pragma unroll
    for (int v = 0; v < 8; ++v) { rp1[v] = 0.f; rp2[v] = 0.f; rp3[v] = 0.f; }
#pragma unroll
    for (int ct = 0; ct < 4; ++ct)
#pragma unroll
      for (int v = 0; v < 8; ++v) {
        float dr = acc[ct][v];
        float c1 = clampf(dr * iq1_v[v] * ik1_c[ct], -0.9f, 0.9f);
        float c2 = clampf(dr * iq2_v[v] * ik2_c[ct], -0.9f, 0.9f);
        float cv = clampf(covscale * (dr - qkm_v[v] - qm_v[v] * ks_c[ct] + qm_v[v] * kms), -10.f, 10.f);
        float mar = clampf(0.01f - c2, 0.f, 5.f);
        p0 += c1; p1 += c1 * c1; p2 += cv; p3 += cv * cv; p4 += c1 * cv;
        rp1[v] += c1; rp2[v] += cv; rp3[v] += mar;
      }
#pragma unroll
    for (int v = 0; v < 8; ++v) {
      float a1 = rp1[v], a2 = rp2[v], a3 = rp3[v];
#pragma unroll
      for (int m = 8; m >= 1; m >>= 1) {
        a1 += __shfl_xor(a1, m); a2 += __shfl_xor(a2, m); a3 += __shfl_xor(a3, m);
      }
      if (cl == 0) {
        int rl = wave * 16 + half * 8 + v;
        rs_c1[rl] += a1; rs_cov[rl] += a2; rs_mar[rl] += a3;
      }
    }
    __syncthreads();
  }
  float ps[5] = {p0, p1, p2, p3, p4};
#pragma unroll
  for (int i = 0; i < 5; ++i) {
    float x = ps[i];
#pragma unroll
    for (int m = 16; m >= 1; m >>= 1) x += __shfl_xor(x, m);
    if (lane == 0) red[wave][i] = x;
  }
  __syncthreads();
  if (tid < 5) {
    float t = 0.f; for (int w = 0; w < 8; ++w) t += red[w][tid];
    atomicAdd(&hsum[h * 16 + tid], t);
  }
  if (tid < 128) {
    float rv = rs_mar[tid] * (1.f / 512.f);
    varrow[hq * 512 + n0 + tid] = rv;
    atomicAdd(&acc4[0], rv);
    atomicAdd(&acc4[1], rv * rv);
    atomicAdd(&acc4[2], rv * rs_c1[tid]);
    atomicAdd(&acc4[3], rv * rs_cov[tid]);
  }
  __syncthreads();
  if (tid < 4) atomicAdd(&hsum[h * 16 + 5 + tid], acc4[tid]);
}

// ---------------------------------------------------------------- mixture-weight MLP
__global__ void k_mlp(const float* __restrict__ feat,
                      const float* __restrict__ w1, const float* __restrict__ b1,
                      const float* __restrict__ lng, const float* __restrict__ lnb,
                      const float* __restrict__ w2, const float* __restrict__ b2,
                      const float* __restrict__ w3, const float* __restrict__ b3,
                      const float* __restrict__ wtemp, float* __restrict__ wts) {
  __shared__ float fy[128], s1[128], s2[64], rbuf[128];
  int tid = threadIdx.x;  // 128 threads
  for (int h = 0; h < 8; ++h) {
    fy[tid] = feat[h * 128 + tid] * (1.f / 8192.f);
    __syncthreads();
    float y = b1[tid];
    for (int i = 0; i < 128; ++i) y += fy[i] * w1[i * 128 + tid];
    rbuf[tid] = y; __syncthreads();
    for (int s = 64; s > 0; s >>= 1) { if (tid < s) rbuf[tid] += rbuf[tid + s]; __syncthreads(); }
    float mu = rbuf[0] * (1.f / 128.f);
    __syncthreads();
    float dv = y - mu; rbuf[tid] = dv * dv; __syncthreads();
    for (int s = 64; s > 0; s >>= 1) { if (tid < s) rbuf[tid] += rbuf[tid + s]; __syncthreads(); }
    float var = rbuf[0] * (1.f / 128.f);
    __syncthreads();
    float ln = (y - mu) * rsqrtf(var + 1e-5f) * lng[tid] + lnb[tid];
    s1[tid] = fmaxf(ln, 0.f);
    __syncthreads();
    if (tid < 64) {
      float z = b2[tid];
      for (int i = 0; i < 128; ++i) z += s1[i] * w2[i * 64 + tid];
      s2[tid] = fmaxf(z, 0.f);
    }
    __syncthreads();
    if (tid == 0) {
      float lg[3];
      for (int j = 0; j < 3; ++j) lg[j] = b3[j];
      for (int i = 0; i < 64; ++i) { float s = s2[i]; for (int j = 0; j < 3; ++j) lg[j] += s * w3[i * 3 + j]; }
      float mx = fmaxf(lg[0], fmaxf(lg[1], lg[2]));
      float e0 = __expf(lg[0] - mx), e1 = __expf(lg[1] - mx), e2 = __expf(lg[2] - mx);
      float inv = 1.f / (e0 + e1 + e2);
      float wl0 = e0 * inv, wl1 = e1 * inv, wl2 = e2 * inv;
      float wt = clampf(wtemp[0], 0.1f, 2.f);
      float t0 = wl0 / wt, t1 = wl1 / wt, t2 = wl2 / wt;
      mx = fmaxf(t0, fmaxf(t1, t2));
      e0 = __expf(t0 - mx); e1 = __expf(t1 - mx); e2 = __expf(t2 - mx);
      inv = 1.f / (e0 + e1 + e2);
      float q0 = clampf(e0 * inv, 0.05f, 0.8f);
      float q1 = clampf(e1 * inv, 0.05f, 0.8f);
      float q2 = clampf(e2 * inv, 0.05f, 0.8f);
      float sum = q0 + q1 + q2;
      wts[h * 4 + 0] = q0 / sum; wts[h * 4 + 1] = q1 / sum; wts[h * 4 + 2] = q2 / sum;
    }
    __syncthreads();
  }
}

// ---------------------------------------------------------------- finalize: stds -> coefficients + temperature
__global__ void k_fin(const float* __restrict__ hsum, const float* __restrict__ wts,
                      float* __restrict__ coef) {
  if (threadIdx.x != 0 || blockIdx.x != 0) return;
  const double Nc = 33554432.0;   // 8*16*512*512
  double T0 = 0, T1 = 0, T2 = 0, T3 = 0, T5 = 0, T6 = 0;
  for (int h = 0; h < 8; ++h) {
    T0 += hsum[h * 16 + 0]; T1 += hsum[h * 16 + 1];
    T2 += hsum[h * 16 + 2]; T3 += hsum[h * 16 + 3];
    T5 += hsum[h * 16 + 5]; T6 += hsum[h * 16 + 6];
  }
  double std1 = sqrt(fmax((T1 - T0 * T0 / Nc) / (Nc - 1.0), 0.0));
  double std2 = sqrt(fmax((T3 - T2 * T2 / Nc) / (Nc - 1.0), 0.0));
  double sv = 512.0 * T5, sv2 = 512.0 * T6;
  double std3 = sqrt(fmax((sv2 - sv * sv / Nc) / (Nc - 1.0), 0.0));
  double Sd = 0, Sd2 = 0;
  for (int h = 0; h < 8; ++h) {
    double a = wts[h * 4 + 0] / (std1 + 1e-6);
    double b = 0.5 * wts[h * 4 + 1] / (std2 + 1e-6);
    double c = 0.5 * wts[h * 4 + 2] / (std3 + 1e-6);
    double S0 = hsum[h * 16 + 0], S1 = hsum[h * 16 + 1], S2 = hsum[h * 16 + 2];
    double S3 = hsum[h * 16 + 3], S4 = hsum[h * 16 + 4], S5 = hsum[h * 16 + 5];
    double S6 = hsum[h * 16 + 6], S7 = hsum[h * 16 + 7], S8 = hsum[h * 16 + 8];
    Sd  += a * S0 + b * S2 + c * 512.0 * S5;
    Sd2 += a * a * S1 + b * b * S3 + c * c * 512.0 * S6 + 2.0 * (a * b * S4 + a * c * S7 + b * c * S8);
    coef[h * 4 + 0] = (float)a; coef[h * 4 + 1] = (float)b; coef[h * 4 + 2] = (float)c;
  }
  double ds = sqrt(fmax((Sd2 - Sd * Sd / Nc) / (Nc - 1.0), 0.0));
  double temp = (ds < 1e-4) ? 0.1 : ((ds < 0.01) ? 0.3 : 0.5 + ds);
  temp = fmin(fmax(temp, 0.1), 3.0);
  coef[32] = (float)(1.0 / temp);
}

// ---------------------------------------------------------------- attention: online softmax + P@V via WMMA
__global__ void k_attn(const _Float16* __restrict__ f16buf,
                       const float* __restrict__ ksum, const float* __restrict__ ik1, const float* __restrict__ ik2,
                       const float* __restrict__ qm, const float* __restrict__ qkm,
                       const float* __restrict__ iq1, const float* __restrict__ iq2,
                       const float* __restrict__ kmsum, const float* __restrict__ varrow,
                       const float* __restrict__ coef, _Float16* __restrict__ hout) {
  __shared__ _Float16 Qs[128][72];
  __shared__ _Float16 Ks[64][72];      // [m][d]
  __shared__ _Float16 Vst[64][72];     // transposed V tile: [d][m]
  __shared__ _Float16 Ps[128][72];     // probabilities [n][m]
  int blk = blockIdx.x, hq = blk >> 2, rb = blk & 3, h = hq >> 4, qb = hq & 15, n0 = rb * 128;
  int tid = threadIdx.x, wave = tid >> 5, lane = tid & 31, half = lane >> 4, cl = lane & 15;
  const _Float16* fq = f16buf + (size_t)hq * NSQ * DHD;
  const _Float16* fk = f16buf + FSZ + (size_t)hq * NSQ * DHD;
  const _Float16* fv = f16buf + 2 * FSZ + (size_t)hq * NSQ * DHD;
#pragma unroll
  for (int it = 0; it < 4; ++it) {
    int pos = (tid + it * 256) * 8; int r = pos >> 6, c = pos & 63;
    *(h8*)&Qs[r][c] = *(const h8*)&fq[(size_t)(n0 + r) * 64 + c];
  }
  int rowbase = hq * 512 + n0 + wave * 16 + half * 8;
  float qm_v[8], qkm_v[8], iq1_v[8], iq2_v[8], vr_v[8];
#pragma unroll
  for (int v = 0; v < 8; ++v) {
    qm_v[v] = qm[rowbase + v]; qkm_v[v] = qkm[rowbase + v];
    iq1_v[v] = iq1[rowbase + v]; iq2_v[v] = iq2[rowbase + v];
    vr_v[v] = varrow[rowbase + v];
  }
  float kms = kmsum[hq];
  float ca = coef[h * 4 + 0], cb = coef[h * 4 + 1], cc = coef[h * 4 + 2], tinv = coef[32];
  const float covscale = (1e-3f / 512.f) / (8.f + 1e-6f);
  float rmax[8], rsum[8];
#pragma unroll
  for (int v = 0; v < 8; ++v) { rmax[v] = -1e30f; rsum[v] = 0.f; }
  __syncthreads();
  // ---- phase 1: online row max / sum (wave-private rows; no cross-wave row state)
  for (int mc = 0; mc < 8; ++mc) {
    int m0 = mc * 64;
#pragma unroll
    for (int it = 0; it < 2; ++it) {
      int pos = (tid + it * 256) * 8; int r = pos >> 6, c = pos & 63;
      *(h8*)&Ks[r][c] = *(const h8*)&fk[(size_t)(m0 + r) * 64 + c];
    }
    __syncthreads();
    float ks_c[4], ik1_c[4], ik2_c[4];
#pragma unroll
    for (int ct = 0; ct < 4; ++ct) {
      int mi = hq * 512 + m0 + ct * 16 + cl;
      ks_c[ct] = ksum[mi]; ik1_c[ct] = ik1[mi]; ik2_c[ct] = ik2[mi];
    }
    v8f acc[4];
#pragma unroll
    for (int ct = 0; ct < 4; ++ct)
#pragma unroll
      for (int i = 0; i < 8; ++i) acc[ct][i] = 0.f;
#pragma unroll
    for (int kk = 0; kk < 2; ++kk) {
      v16h a = frag_rowmajor(&Qs[0][0], 72, wave * 16, kk * 32, lane);
#pragma unroll
      for (int ct = 0; ct < 4; ++ct) {
        v16h b = frag_rowmajor(&Ks[0][0], 72, ct * 16, kk * 32, lane);
        acc[ct] = wmma_f16(a, b, acc[ct]);
      }
    }
#pragma unroll
    for (int ct = 0; ct < 4; ++ct)
#pragma unroll
      for (int v = 0; v < 8; ++v) {
        float dr = acc[ct][v];
        float c1 = clampf(dr * iq1_v[v] * ik1_c[ct], -0.9f, 0.9f);
        float cv = clampf(covscale * (dr - qkm_v[v] - qm_v[v] * ks_c[ct] + qm_v[v] * kms), -10.f, 10.f);
        acc[ct][v] = tinv * (ca * c1 + cb * cv + cc * vr_v[v]);
      }
#pragma unroll
    for (int v = 0; v < 8; ++v) {
      float cm = fmaxf(fmaxf(acc[0][v], acc[1][v]), fmaxf(acc[2][v], acc[3][v]));
#pragma unroll
      for (int m = 8; m >= 1; m >>= 1) cm = fmaxf(cm, __shfl_xor(cm, m));
      float mnew = fmaxf(rmax[v], cm);
      float pe = __expf(acc[0][v] - mnew) + __expf(acc[1][v] - mnew)
               + __expf(acc[2][v] - mnew) + __expf(acc[3][v] - mnew);
#pragma unroll
      for (int m = 8; m >= 1; m >>= 1) pe += __shfl_xor(pe, m);
      rsum[v] = rsum[v] * __expf(rmax[v] - mnew) + pe;
      rmax[v] = mnew;
    }
    __syncthreads();
  }
  // ---- phase 2: recompute, exp -> LDS, P@V via WMMA
  v8f oacc[4];
#pragma unroll
  for (int ct = 0; ct < 4; ++ct)
#pragma unroll
    for (int i = 0; i < 8; ++i) oacc[ct][i] = 0.f;
  for (int mc = 0; mc < 8; ++mc) {
    int m0 = mc * 64;
#pragma unroll
    for (int it = 0; it < 2; ++it) {
      int pos = (tid + it * 256) * 8; int r = pos >> 6, c = pos & 63;  // r = m, c = d-base
      *(h8*)&Ks[r][c] = *(const h8*)&fk[(size_t)(m0 + r) * 64 + c];
      h8 vv = *(const h8*)&fv[(size_t)(m0 + r) * 64 + c];
#pragma unroll
      for (int e = 0; e < 8; ++e) Vst[c + e][r] = vv[e];   // transpose to [d][m]
    }
    __syncthreads();
    float ks_c[4], ik1_c[4], ik2_c[4];
#pragma unroll
    for (int ct = 0; ct < 4; ++ct) {
      int mi = hq * 512 + m0 + ct * 16 + cl;
      ks_c[ct] = ksum[mi]; ik1_c[ct] = ik1[mi]; ik2_c[ct] = ik2[mi];
    }
    v8f acc[4];
#pragma unroll
    for (int ct = 0; ct < 4; ++ct)
#pragma unroll
      for (int i = 0; i < 8; ++i) acc[ct][i] = 0.f;
#pragma unroll
    for (int kk = 0; kk < 2; ++kk) {
      v16h a = frag_rowmajor(&Qs[0][0], 72, wave * 16, kk * 32, lane);
#pragma unroll
      for (int ct = 0; ct < 4; ++ct) {
        v16h b = frag_rowmajor(&Ks[0][0], 72, ct * 16, kk * 32, lane);
        acc[ct] = wmma_f16(a, b, acc[ct]);
      }
    }
#pragma unroll
    for (int ct = 0; ct < 4; ++ct)
#pragma unroll
      for (int v = 0; v < 8; ++v) {
        float dr = acc[ct][v];
        float c1 = clampf(dr * iq1_v[v] * ik1_c[ct], -0.9f, 0.9f);
        float cv = clampf(covscale * (dr - qkm_v[v] - qm_v[v] * ks_c[ct] + qm_v[v] * kms), -10.f, 10.f);
        float x = tinv * (ca * c1 + cb * cv + cc * vr_v[v]);
        Ps[wave * 16 + half * 8 + v][ct * 16 + cl] = (_Float16)__expf(x - rmax[v]);
      }
    __syncthreads();
#pragma unroll
    for (int kk = 0; kk < 2; ++kk) {
      v16h a = frag_rowmajor(&Ps[0][0], 72, wave * 16, kk * 32, lane);   // wave-private rows
#pragma unroll
      for (int ct = 0; ct < 4; ++ct) {
        v16h b = frag_rowmajor(&Vst[0][0], 72, ct * 16, kk * 32, lane);  // [d][m] -> B[k=m][n=d]
        oacc[ct] = wmma_f16(a, b, oacc[ct]);
      }
    }
    __syncthreads();
  }
#pragma unroll
  for (int ct = 0; ct < 4; ++ct)
#pragma unroll
    for (int v = 0; v < 8; ++v) {
      float o = oacc[ct][v] / rsum[v];
      int n = n0 + wave * 16 + half * 8 + v;
      int cg = h * 64 + ct * 16 + cl;
      hout[(size_t)(qb * 512 + n) * 512 + cg] = (_Float16)o;
    }
}

// ---------------------------------------------------------------- output GEMM: hout @ W_out + b_out
__global__ void k_gemm_out(const _Float16* __restrict__ A16, const _Float16* __restrict__ w16,
                           const float* __restrict__ bout, float* __restrict__ out) {
  __shared__ _Float16 As[128][72];
  __shared__ _Float16 Bst[64][72];     // transposed weight tile: [n][k]
  int tid = threadIdx.x, wave = tid >> 5, lane = tid & 31;
  int wr = wave & 3, wc = wave >> 2;
  int row0 = blockIdx.x * 128, col0 = blockIdx.y * 64;
  v8f acc[2][2];
#pragma unroll
  for (int a = 0; a < 2; ++a)
#pragma unroll
    for (int b = 0; b < 2; ++b)
#pragma unroll
      for (int i = 0; i < 8; ++i) acc[a][b][i] = 0.f;
  for (int kb = 0; kb < 512; kb += 64) {
#pragma unroll
    for (int it = 0; it < 4; ++it) {
      int pos = (tid + it * 256) * 8; int r = pos >> 6, c = pos & 63;
      *(h8*)&As[r][c] = *(const h8*)&A16[(size_t)(row0 + r) * 512 + kb + c];
    }
#pragma unroll
    for (int it = 0; it < 2; ++it) {
      int pos = (tid + it * 256) * 8; int r = pos >> 6, c = pos & 63;
      h8 bv = *(const h8*)&w16[(size_t)(kb + r) * 512 + col0 + c];
#pragma unroll
      for (int e = 0; e < 8; ++e) Bst[c + e][r] = bv[e];
    }
    __syncthreads();
#pragma unroll
    for (int kk = 0; kk < 2; ++kk) {
      v16h af[2], bf[2];
#pragma unroll
      for (int rt = 0; rt < 2; ++rt) af[rt] = frag_rowmajor(&As[0][0], 72, wr * 32 + rt * 16, kk * 32, lane);
#pragma unroll
      for (int ct = 0; ct < 2; ++ct) bf[ct] = frag_rowmajor(&Bst[0][0], 72, wc * 32 + ct * 16, kk * 32, lane);
#pragma unroll
      for (int rt = 0; rt < 2; ++rt)
#pragma unroll
        for (int ct = 0; ct < 2; ++ct) acc[rt][ct] = wmma_f16(af[rt], bf[ct], acc[rt][ct]);
    }
    __syncthreads();
  }
  int half = lane >> 4, cl = lane & 15;
#pragma unroll
  for (int rt = 0; rt < 2; ++rt)
#pragma unroll
    for (int ct = 0; ct < 2; ++ct)
#pragma unroll
      for (int v = 0; v < 8; ++v) {
        int rg = row0 + wr * 32 + rt * 16 + half * 8 + v;
        int cg = col0 + wc * 32 + ct * 16 + cl;
        out[(size_t)rg * 512 + cg] = acc[rt][ct][v] + bout[cg];
      }
}

// ---------------------------------------------------------------- host-side launch
extern "C" void kernel_launch(void* const* d_in, const int* in_sizes, int n_in,
                              void* d_out, int out_size, void* d_ws, size_t ws_size,
                              hipStream_t stream) {
  (void)in_sizes; (void)n_in; (void)out_size; (void)ws_size;
  const float* q      = (const float*)d_in[0];
  const float* k      = (const float*)d_in[1];
  const float* v      = (const float*)d_in[2];
  const float* ln1_g  = (const float*)d_in[3];
  const float* ln1_b  = (const float*)d_in[4];
  const float* W_in   = (const float*)d_in[5];
  const float* wp_w1  = (const float*)d_in[6];
  const float* wp_b1  = (const float*)d_in[7];
  const float* wp_lng = (const float*)d_in[8];
  const float* wp_lnb = (const float*)d_in[9];
  const float* wp_w2  = (const float*)d_in[10];
  const float* wp_b2  = (const float*)d_in[11];
  const float* wp_w3  = (const float*)d_in[12];
  const float* wp_b3  = (const float*)d_in[13];
  const float* wtemp  = (const float*)d_in[14];
  const float* W_out  = (const float*)d_in[15];
  const float* b_out  = (const float*)d_in[16];
  float* out = (float*)d_out;

  char* w = (char*)d_ws;
  auto take = [&](size_t bytes) -> void* {
    void* p = (void*)w;
    w += (bytes + 255) & ~((size_t)255);
    return p;
  };
  _Float16* XLN    = (_Float16*)take((size_t)TROWS * DIMX * 2);
  _Float16* F16    = (_Float16*)take(3 * FSZ * 2);
  _Float16* HOUT   = (_Float16*)take((size_t)MROWS * DIMX * 2);
  _Float16* WIN16  = (_Float16*)take((size_t)512 * 512 * 2);
  _Float16* WOUT16 = (_Float16*)take((size_t)512 * 512 * 2);
  float* VARROW = (float*)take((size_t)65536 * 4);
  float* QM     = (float*)take((size_t)65536 * 4);
  float* QKM    = (float*)take((size_t)65536 * 4);
  float* IQ1    = (float*)take((size_t)65536 * 4);
  float* IQ2    = (float*)take((size_t)65536 * 4);
  float* KSUM   = (float*)take((size_t)65536 * 4);
  float* IK1    = (float*)take((size_t)65536 * 4);
  float* IK2    = (float*)take((size_t)65536 * 4);
  float* KMSUM  = (float*)take((size_t)128 * 4);
  float* FEAT   = (float*)take((size_t)1024 * 4);
  float* HSUM   = (float*)take((size_t)128 * 4);
  float* WTS    = (float*)take((size_t)64 * 4);
  float* COEF   = (float*)take((size_t)64 * 4);

  k_init<<<1, 256, 0, stream>>>(FEAT, HSUM, COEF, WTS);
  k_cvt<<<1024, 256, 0, stream>>>(W_in, W_out, WIN16, WOUT16);
  k_ln<<<TROWS / 8, 256, 0, stream>>>(q, k, v, ln1_g, ln1_b, XLN);
  k_gemm_in<<<dim3(192, 8), 256, 0, stream>>>(XLN, WIN16, F16);
  k_stats2<<<128, 256, 0, stream>>>(F16, KSUM, IK1, IK2, QM, QKM, IQ1, IQ2, KMSUM, FEAT);
  k_scores<<<512, 256, 0, stream>>>(F16, KSUM, IK1, IK2, QM, QKM, IQ1, IQ2, KMSUM, VARROW, HSUM);
  k_mlp<<<1, 128, 0, stream>>>(FEAT, wp_w1, wp_b1, wp_lng, wp_lnb, wp_w2, wp_b2, wp_w3, wp_b3, wtemp, WTS);
  k_fin<<<1, 1, 0, stream>>>(HSUM, WTS, COEF);
  k_attn<<<512, 256, 0, stream>>>(F16, KSUM, IK1, IK2, QM, QKM, IQ1, IQ2, KMSUM, VARROW, COEF, HOUT);
  k_gemm_out<<<dim3(64, 8), 256, 0, stream>>>(HOUT, WOUT16, b_out, out);
}
